// GNNModel_55035710931256
// MI455X (gfx1250) — compile-verified, matrix operand order checked by reading, not run on hardware
//
#include <hip/hip_runtime.h>

// Problem constants (from reference)
#define NU   50000
#define NPR  20000
#define DIN  128      // DU == DP
#define HID  256
#define OUTD 128
#define NEDGE 800000
#define NPAIR 200000

typedef __attribute__((ext_vector_type(16))) __bf16          v16bf;
typedef __attribute__((ext_vector_type(16))) unsigned short  v16us;
typedef __attribute__((ext_vector_type(8)))  float           v8f;

__device__ __forceinline__ unsigned short f2bf(float f) {
  unsigned int u = __builtin_bit_cast(unsigned int, f);
  u += 0x7FFFu + ((u >> 16) & 1u);          // round-to-nearest-even
  return (unsigned short)(u >> 16);
}

__device__ __forceinline__ unsigned int pack2bf(float lo, float hi) {
  return (unsigned int)f2bf(lo) | ((unsigned int)f2bf(hi) << 16);
}

// -------------------------------------------------------------------------
// Generic GEMM: C[M,N] = epi(A[M,K] @ W[K,N] (+bias) (+C) (relu))
// Block: 256 threads = 8 waves. Block tile 64(M) x 128(N).
// Wave grid 2x4; each wave computes a 32x32 tile = 2x2 WMMA 16x16 tiles.
// K stepped by 32 using v_wmma_f32_16x16x32_bf16.
// A tile stored row-major bf16 in LDS; B tile stored TRANSPOSED (col-major)
// so every fragment load is two contiguous 16B runs -> ds_load_b128.
// Requires: N % 128 == 0, K % 32 == 0 (true: N in {256,128}, K in {128,256}).
// -------------------------------------------------------------------------
__global__ __launch_bounds__(256)
void gemm_bf16_wmma(const float* __restrict__ A, const float* __restrict__ W,
                    const float* __restrict__ bias, float* __restrict__ C,
                    int M, int N, int K, int relu, int addC) {
  __shared__ unsigned short As [64][40];   // A tile:  [m][k]  bf16 (+pad)
  __shared__ unsigned short BsT[128][40];  // B tile:  [n][k]  bf16 (+pad), transposed

  const int tid  = threadIdx.x;
  const int lane = tid & 31;
  const int wave = tid >> 5;
  const int m0   = blockIdx.x * 64;
  const int n0   = blockIdx.y * 128;
  const int wmb  = (wave >> 2) * 32;       // wave M base in tile
  const int wnb  = (wave & 3) * 32;        // wave N base in tile

  v8f acc[2][2] = {};

  for (int k0 = 0; k0 < K; k0 += 32) {
    // --- stage A: 64x32 fp32 -> bf16 LDS (2 float4 per thread, coalesced)
#pragma unroll
    for (int i = 0; i < 2; ++i) {
      int idx = tid + i * 256;             // 0..511 float4 slots
      int row = idx >> 3;                  // 8 float4 per 32-wide row
      int c4  = (idx & 7) << 2;
      float4 v = make_float4(0.f, 0.f, 0.f, 0.f);
      if (m0 + row < M)
        v = *(const float4*)(A + (size_t)(m0 + row) * K + (k0 + c4));
      *(unsigned int*)&As[row][c4 + 0] = pack2bf(v.x, v.y);
      *(unsigned int*)&As[row][c4 + 2] = pack2bf(v.z, v.w);
    }
    // --- stage B transposed: 32x128 fp32 -> BsT[n][k] bf16.
    // Each slot covers 2 adjacent k-rows x 4 cols => 4 dword LDS stores.
#pragma unroll
    for (int i = 0; i < 2; ++i) {
      int idx = tid + i * 256;             // 0..511 slots
      int cg  = (idx & 31) << 2;           // column group (4 cols)
      int r   = (idx >> 5) << 1;           // even k row
      float4 v0 = *(const float4*)(W + (size_t)(k0 + r)     * N + (n0 + cg));
      float4 v1 = *(const float4*)(W + (size_t)(k0 + r + 1) * N + (n0 + cg));
      *(unsigned int*)&BsT[cg + 0][r] = pack2bf(v0.x, v1.x);
      *(unsigned int*)&BsT[cg + 1][r] = pack2bf(v0.y, v1.y);
      *(unsigned int*)&BsT[cg + 2][r] = pack2bf(v0.z, v1.z);
      *(unsigned int*)&BsT[cg + 3][r] = pack2bf(v0.w, v1.w);
    }
    // prefetch next K-step tiles while this step computes
    if (k0 + 32 < K) {
      int arow = m0 + (tid >> 3);
      if (arow < M) __builtin_prefetch(A + (size_t)arow * K + (k0 + 32), 0, 1);
      __builtin_prefetch(W + (size_t)(k0 + 32 + (tid >> 5)) * N + n0 + ((tid & 31) << 2), 0, 1);
    }
    __syncthreads();

    // --- build fragments per ISA 16-bit layout:
    // lane l, element e -> K = (e&7) + (e>>3)*16 + (l>=16 ? 8 : 0)
    // Both A and B(T) are [major][k], so each fragment = 2 contiguous 16B runs.
    v16bf afrag[2], bfrag[2];
#pragma unroll
    for (int mt = 0; mt < 2; ++mt) {
      int row = wmb + mt * 16 + (lane & 15);
      v16us u;
#pragma unroll
      for (int e = 0; e < 16; ++e) {
        int kk = (e & 7) + ((e >> 3) << 4) + ((lane >> 4) << 3);
        u[e] = As[row][kk];
      }
      afrag[mt] = __builtin_bit_cast(v16bf, u);
    }
#pragma unroll
    for (int nt = 0; nt < 2; ++nt) {
      int col = wnb + nt * 16 + (lane & 15);
      v16us u;
#pragma unroll
      for (int e = 0; e < 16; ++e) {
        int kk = (e & 7) + ((e >> 3) << 4) + ((lane >> 4) << 3);
        u[e] = BsT[col][kk];
      }
      bfrag[nt] = __builtin_bit_cast(v16bf, u);
    }

#pragma unroll
    for (int mt = 0; mt < 2; ++mt)
#pragma unroll
      for (int nt = 0; nt < 2; ++nt)
        acc[mt][nt] = __builtin_amdgcn_wmma_f32_16x16x32_bf16(
            false, afrag[mt], false, bfrag[nt],
            (short)0, acc[mt][nt], false, false);
    __syncthreads();
  }

  // --- epilogue: C/D layout -> lanes 0-15: M=r, lanes 16-31: M=r+8; N=lane&15
#pragma unroll
  for (int mt = 0; mt < 2; ++mt) {
#pragma unroll
    for (int nt = 0; nt < 2; ++nt) {
      int col = n0 + wnb + nt * 16 + (lane & 15);
      float bv = bias ? bias[col] : 0.0f;
#pragma unroll
      for (int r = 0; r < 8; ++r) {
        int row = m0 + wmb + mt * 16 + r + ((lane >> 4) << 3);
        if (row < M) {
          size_t off = (size_t)row * N + col;
          float v = acc[mt][nt][r] + bv;
          if (addC) v += C[off];
          if (relu) v = fmaxf(v, 0.0f);
          C[off] = v;
        }
      }
    }
  }
}

// -------------------------------------------------------------------------
// Helper kernels
// -------------------------------------------------------------------------
__global__ void zero_f32(float* __restrict__ p, long long n) {
  long long i = (long long)blockIdx.x * blockDim.x + threadIdx.x;
  long long stride = (long long)gridDim.x * blockDim.x;
  for (; i < n; i += stride) p[i] = 0.0f;
}

__global__ void deg_count(const int* __restrict__ dst, float* __restrict__ deg, int E) {
  int e = blockIdx.x * blockDim.x + threadIdx.x;
  if (e < E) atomicAdd(&deg[dst[e]], 1.0f);
}

// one thread per (edge, 4-float chunk); D multiple of 4
__global__ void scatter_add4(const float* __restrict__ m, const int* __restrict__ src,
                             const int* __restrict__ dst, float* __restrict__ acc,
                             int E, int D) {
  long long t = (long long)blockIdx.x * blockDim.x + threadIdx.x;
  int chunks = D >> 2;
  long long total = (long long)E * chunks;
  if (t >= total) return;
  int e = (int)(t / chunks);
  int c = (int)(t % chunks) << 2;
  float4 v = *(const float4*)(m + (size_t)src[e] * D + c);
  float* a = acc + (size_t)dst[e] * D + c;
  atomicAdd(a + 0, v.x);
  atomicAdd(a + 1, v.y);
  atomicAdd(a + 2, v.z);
  atomicAdd(a + 3, v.w);
}

__global__ void div_deg(float* __restrict__ acc, const float* __restrict__ deg,
                        int n, int D) {
  long long t = (long long)blockIdx.x * blockDim.x + threadIdx.x;
  long long total = (long long)n * D;
  if (t >= total) return;
  int row = (int)(t / D);
  acc[t] = acc[t] / fmaxf(deg[row], 1.0f);
}

// one wave (32 lanes) per row; 8 rows per 256-thread block
__global__ __launch_bounds__(256)
void rownorm(const float* __restrict__ z, float* __restrict__ out, int n, int D) {
  int wave = threadIdx.x >> 5;
  int lane = threadIdx.x & 31;
  int row  = blockIdx.x * 8 + wave;
  if (row >= n) return;
  const float* zr = z + (size_t)row * D;
  float s = 0.0f;
  for (int j = lane; j < D; j += 32) { float v = zr[j]; s += v * v; }
#pragma unroll
  for (int o = 16; o > 0; o >>= 1) s += __shfl_xor(s, o, 32);
  float nrm = sqrtf(s);
  float inv = (nrm == 0.0f) ? 1.0f : (1.0f / nrm);
  for (int j = lane; j < D; j += 32) out[(size_t)row * D + j] = zr[j] * inv;
}

// one wave per pair: out[p] = dot(u[iu[p]], v[ip[p]]) over D floats
__global__ __launch_bounds__(256)
void pair_dot(const float* __restrict__ u, const float* __restrict__ v,
              const int* __restrict__ iu, const int* __restrict__ ip,
              float* __restrict__ out, int P, int D) {
  int wave = threadIdx.x >> 5;
  int lane = threadIdx.x & 31;
  int p = blockIdx.x * 8 + wave;
  if (p >= P) return;
  const float* a = u + (size_t)iu[p] * D;
  const float* b = v + (size_t)ip[p] * D;
  float s = 0.0f;
  for (int j = lane; j < D; j += 32) s += a[j] * b[j];
#pragma unroll
  for (int o = 16; o > 0; o >>= 1) s += __shfl_xor(s, o, 32);
  if (lane == 0) out[p] = s;
}

// -------------------------------------------------------------------------
// Host orchestration
// -------------------------------------------------------------------------
extern "C" void kernel_launch(void* const* d_in, const int* in_sizes, int n_in,
                              void* d_out, int out_size, void* d_ws, size_t ws_size,
                              hipStream_t stream) {
  const float* user_x = (const float*)d_in[0];
  const float* prod_x = (const float*)d_in[1];
  const float* W_ue = (const float*)d_in[2];  const float* b_ue = (const float*)d_in[3];
  const float* W_pe = (const float*)d_in[4];  const float* b_pe = (const float*)d_in[5];
  const float* l1_up_pre = (const float*)d_in[6];
  const float* l1_up_nb  = (const float*)d_in[7];
  const float* l1_up_self= (const float*)d_in[8];
  const float* l1_pu_pre = (const float*)d_in[9];
  const float* l1_pu_nb  = (const float*)d_in[10];
  const float* l1_pu_self= (const float*)d_in[11];
  const float* l2_up_pre = (const float*)d_in[12];
  const float* l2_up_nb  = (const float*)d_in[13];
  const float* l2_up_self= (const float*)d_in[14];
  const float* l2_pu_pre = (const float*)d_in[15];
  const float* l2_pu_nb  = (const float*)d_in[16];
  const float* l2_pu_self= (const float*)d_in[17];
  const int* eu_src = (const int*)d_in[18];
  const int* eu_dst = (const int*)d_in[19];
  const int* ep_src = (const int*)d_in[20];
  const int* ep_dst = (const int*)d_in[21];
  const int* pos_u = (const int*)d_in[22];
  const int* pos_p = (const int*)d_in[23];
  const int* neg_u = (const int*)d_in[24];
  const int* neg_p = (const int*)d_in[25];

  // workspace layout (floats)
  float* ws   = (float*)d_ws;
  float* h_u  = ws;                               // NU  x HID
  float* h_p  = h_u  + (size_t)NU  * HID;         // NPR x HID
  float* hu1  = h_p  + (size_t)NPR * HID;         // NU  x HID
  float* hp1  = hu1  + (size_t)NU  * HID;         // NPR x HID
  float* mbuf = hp1  + (size_t)NPR * HID;         // NU  x HID (messages / z scratch)
  float* agg  = mbuf + (size_t)NU  * HID;         // NU  x HID (segment sums)
  float* deg  = agg  + (size_t)NU  * HID;         // NU

  // outputs
  float* hu2 = (float*)d_out;                     // NU  x OUTD
  float* hp2 = hu2 + (size_t)NU  * OUTD;          // NPR x OUTD
  float* pos = hp2 + (size_t)NPR * OUTD;          // NPAIR
  float* neg = pos + NPAIR;                       // NPAIR

  auto gemm = [&](const float* A, const float* W, const float* b, float* C,
                  int M, int N, int K, int relu, int addC) {
    dim3 g((M + 63) / 64, N / 128);
    gemm_bf16_wmma<<<g, 256, 0, stream>>>(A, W, b, C, M, N, K, relu, addC);
  };
  auto zero = [&](float* p, long long n) {
    int blocks = (int)((n + 255) / 256); if (blocks > 65535) blocks = 65535;
    zero_f32<<<blocks, 256, 0, stream>>>(p, n);
  };
  auto mp = [&](const float* hsrc, int nsrc, const float* hdst, int ndst,
                const int* esrc, const int* edst,
                const float* Wpre, const float* Wself, const float* Wnb,
                int dout, float* result) {
    // m = relu(hsrc @ Wpre)  [nsrc, HID]
    gemm(hsrc, Wpre, nullptr, mbuf, nsrc, HID, HID, 1, 0);
    // segment mean
    zero(agg, (long long)ndst * HID);
    zero(deg, ndst);
    deg_count<<<(NEDGE + 255) / 256, 256, 0, stream>>>(edst, deg, NEDGE);
    {
      long long total = (long long)NEDGE * (HID / 4);
      scatter_add4<<<(int)((total + 255) / 256), 256, 0, stream>>>(
          mbuf, esrc, edst, agg, NEDGE, HID);
    }
    {
      long long total = (long long)ndst * HID;
      div_deg<<<(int)((total + 255) / 256), 256, 0, stream>>>(agg, deg, ndst, HID);
    }
    // z = relu(hdst @ Wself + agg @ Wnb)  [ndst, dout]   (reuse mbuf as z)
    gemm(hdst, Wself, nullptr, mbuf, ndst, dout, HID, 0, 0);
    gemm(agg,  Wnb,   nullptr, mbuf, ndst, dout, HID, 1, 1);
    // result = l2-normalized rows of z
    rownorm<<<(ndst + 7) / 8, 256, 0, stream>>>(mbuf, result, ndst, dout);
  };

  // node embedding projections: h = x @ W + b
  gemm(user_x, W_ue, b_ue, h_u, NU,  HID, DIN, 0, 0);
  gemm(prod_x, W_pe, b_pe, h_p, NPR, HID, DIN, 0, 0);

  // layer 1
  mp(h_u, NU, h_p, NPR, eu_src, eu_dst, l1_up_pre, l1_up_self, l1_up_nb, HID, hp1);
  mp(h_p, NPR, h_u, NU, ep_src, ep_dst, l1_pu_pre, l1_pu_self, l1_pu_nb, HID, hu1);

  // layer 2 (hidden -> out), writes directly into d_out
  mp(hu1, NU, hp1, NPR, eu_src, eu_dst, l2_up_pre, l2_up_self, l2_up_nb, OUTD, hp2);
  mp(hp1, NPR, hu1, NU, ep_src, ep_dst, l2_pu_pre, l2_pu_self, l2_pu_nb, OUTD, hu2);

  // cosine predictions (rows already unit-norm)
  pair_dot<<<(NPAIR + 7) / 8, 256, 0, stream>>>(hu2, hp2, pos_u, pos_p, pos, NPAIR, OUTD);
  pair_dot<<<(NPAIR + 7) / 8, 256, 0, stream>>>(hu2, hp2, neg_u, neg_p, neg, NPAIR, OUTD);
}
